// ActorNetwork_67508295958577
// MI455X (gfx1250) — matrix operand order repaired and checked by reading, not compiled, not run on hardware
//
#include <hip/hip_runtime.h>
#include <hip/hip_bf16.h>

#define GAMES   8
#define BROWS   8192
#define DIN     512
#define HID1    1024
#define HID2    1024
#define AOUT    128
#define MT      128                 // rows per M tile
#define NT_TILES 72                 // 8192/128 + GAMES pad tiles
#define NSLOT   (NT_TILES * MT)     // 9216 padded row slots
#define BN      64                  // block N tile
#define SA      40                  // LDS row stride (bf16 elems): 32 + 8 pad (80B)
#define SB      40

typedef __attribute__((ext_vector_type(16))) __bf16 v16bf;
typedef __attribute__((ext_vector_type(8)))  float  v8f;
typedef unsigned int u32x4 __attribute__((ext_vector_type(4)));
typedef int          i32x8 __attribute__((ext_vector_type(8)));
typedef int          i32x4 __attribute__((ext_vector_type(4)));

union Frag { uint4 q[2]; v16bf v; };

__device__ __forceinline__ unsigned short f2bf(float f) {
  union { float f; unsigned u; } x; x.f = f;
  unsigned r = ((x.u >> 16) & 1u) + 0x7fffu;   // round-to-nearest-even
  return (unsigned short)((x.u + r) >> 16);
}

// ---------------------------------------------------------------------------
// Scheduling kernel: stable counting sort by game + padded tile schedule.
// ---------------------------------------------------------------------------
__global__ __launch_bounds__(256)
void build_schedule(const int* __restrict__ idx,
                    int* __restrict__ sched,      // [NSLOT] slot -> original row (-1 pad)
                    int* __restrict__ tile_game,  // [NT_TILES]
                    int* __restrict__ l3_in,      // [NSLOT] slot -> h2 source slot (-1 pad)
                    int* __restrict__ tpos)       // [BROWS] tight pos -> slot
{
  __shared__ int cnt[256][GAMES];
  __shared__ int tot[GAMES];
  __shared__ int tstart[GAMES];
  __shared__ int tbase[GAMES];
  const int t = threadIdx.x;
  const int CH = BROWS / 256;       // 32

  int loc[GAMES];
  #pragma unroll
  for (int g = 0; g < GAMES; ++g) loc[g] = 0;
  for (int i = t * CH; i < t * CH + CH; ++i) loc[idx[i]]++;
  #pragma unroll
  for (int g = 0; g < GAMES; ++g) cnt[t][g] = loc[g];
  __syncthreads();

  if (t < GAMES) {
    int run = 0;
    for (int c = 0; c < 256; ++c) { int v = cnt[c][t]; cnt[c][t] = run; run += v; }
    tot[t] = run;
  }
  __syncthreads();

  if (t == 0) {
    int ts = 0, tb = 0;
    for (int g = 0; g < GAMES; ++g) {
      tstart[g] = ts; tbase[g] = tb;
      ts += tot[g];
      int nt = (tot[g] + MT - 1) / MT;
      for (int k = 0; k < nt; ++k) tile_game[tb + k] = g;
      tb += nt;
    }
    for (int T = tb; T < NT_TILES; ++T) tile_game[T] = 0;
  }
  for (int s = t; s < NSLOT; s += 256) sched[s] = -1;
  __syncthreads();

  #pragma unroll
  for (int g = 0; g < GAMES; ++g) loc[g] = cnt[t][g];
  for (int i = t * CH; i < t * CH + CH; ++i) {
    int g = idx[i];
    int rank = loc[g]++;
    int slot = tbase[g] * MT + rank;
    sched[slot] = i;
    tpos[tstart[g] + rank] = slot;
  }
  __syncthreads();

  for (int s = t; s < NSLOT; s += 256) {
    int b = sched[s];
    l3_in[s] = (b >= 0) ? tpos[b] : -1;
  }
}

// ---------------------------------------------------------------------------
// bf16 WMMA GEMM, double-buffered LDS.
// ASTAGE: 0 = f32 source, convert in VGPRs (layer 1)
//         1 = bf16 contiguous source via TDM tensor_load_to_lds (layer 2)
//         2 = bf16 gathered source via global_load_async_to_lds_b128 (layer 3)
// ACT: 0 = relu, 1 = tanh.  OUT_BF16: bf16 direct store vs f32 scatter store.
// ---------------------------------------------------------------------------
template<int KD, int ASTAGE, int ACT, bool OUT_BF16>
__global__ __launch_bounds__(256)
void gemm_wmma(const void* __restrict__ Asrc, int lda,
               const float* __restrict__ W, long wGameStride,
               const float* __restrict__ bias, int bGameStride,
               void* __restrict__ Out, int ldo, int Ntotal,
               const int* __restrict__ gatherA,
               const int* __restrict__ tileGame,
               const int* __restrict__ scatterOut)
{
  __shared__ unsigned short alds[2][MT * SA];   // 2 x 10240 B
  __shared__ unsigned short blds[2][BN * SB];   // 2 x  5120 B

  const int tid   = threadIdx.x;
  const int mtile = blockIdx.x;
  const int n0    = blockIdx.y * BN;
  const int slot0 = mtile * MT;
  const int g     = tileGame ? tileGame[mtile] : 0;
  const float* Wg = W + (long)g * wGameStride;
  const float* bg = bias + (long)g * bGameStride;

  // A staging: 2 threads per row, 16 elems each. Pad rows clamp to row 0:
  // their garbage only affects pad output rows, which are never stored.
  const int ar = tid >> 1;
  const int ac = (tid & 1) * 16;
  int arow = gatherA ? gatherA[slot0 + ar] : (slot0 + ar);
  arow = arow < 0 ? 0 : arow;

  // B staging: 8 threads per k-row, 8 columns each (transpose into LDS)
  const int bkr = tid >> 3;
  const int bnc = (tid & 7) * 8;

  const int wave  = tid >> 5;
  const int lane  = tid & 31;
  const int waveM = (wave >> 1) * 32;
  const int waveN = (wave & 1) * 32;
  const int lrow  = lane & 15;
  const int akoff = (lane >> 4) * 8;    // A frag: lane<16 -> K{0..7,16..23}
  const int bkoff = (lane >> 4) * 8;    // B frag: lane<16 -> K{0..7,16..23}

  auto stageA = [&](int k0, int buf) {
    if (ASTAGE == 0) {
      const float* As = (const float*)Asrc;
      const float4* p = (const float4*)(As + (long)arow * lda + k0 + ac);
      float4 f0 = p[0], f1 = p[1], f2 = p[2], f3 = p[3];
      float fv[16] = { f0.x,f0.y,f0.z,f0.w, f1.x,f1.y,f1.z,f1.w,
                       f2.x,f2.y,f2.z,f2.w, f3.x,f3.y,f3.z,f3.w };
      #pragma unroll
      for (int j = 0; j < 4; ++j) {
        uint2 pk;
        pk.x = (unsigned)f2bf(fv[4*j+0]) | ((unsigned)f2bf(fv[4*j+1]) << 16);
        pk.y = (unsigned)f2bf(fv[4*j+2]) | ((unsigned)f2bf(fv[4*j+3]) << 16);
        *(uint2*)&alds[buf][ar * SA + ac + j * 4] = pk;
      }
    } else if (ASTAGE == 1) {
#if __has_builtin(__builtin_amdgcn_tensor_load_to_lds)
      if (wave == 0) {
        // TDM D#: 2D bf16 tile [tile_dim0=32 x tile_dim1=128], row stride lda,
        // hardware pad 4 DWORDs every 16 DWORDs -> LDS row stride 80B == SA.
        const unsigned short* As = (const unsigned short*)Asrc;
        unsigned long long ga = (unsigned long long)(uintptr_t)
                                (As + (long)slot0 * lda + k0);
        unsigned aoff = (unsigned)(uintptr_t)&alds[buf][0];
        u32x4 g0;
        g0[0] = 1u;                                        // count=1, user desc
        g0[1] = aoff;                                      // lds_addr
        g0[2] = (unsigned)(ga & 0xffffffffu);              // global_addr lo
        g0[3] = (unsigned)((ga >> 32) & 0x01ffffffu) | (2u << 30); // hi | type=2
        const unsigned td0 = 32u, td1 = (unsigned)MT;
        i32x8 g1;
        g1[0] = (int)((1u << 16)      // data_size = 2B
                    | (1u << 20)      // pad_enable
                    | (3u << 22)      // pad_interval: 16 DWORDs
                    | (3u << 25));    // pad_amount: 4 DWORDs
        g1[1] = (int)((td0 & 0xffffu) << 16);              // tensor_dim0 lo16
        g1[2] = (int)((td0 >> 16) | ((td1 & 0xffffu) << 16));
        g1[3] = (int)((td1 >> 16) | (32u << 16));          // tile_dim0 = 32
        g1[4] = (int)(unsigned)MT;                         // tile_dim1=128, dim2=0
        g1[5] = (int)(unsigned)lda;                        // dim0_stride lo32
        g1[6] = 0;
        g1[7] = 0;
        i32x4 z4 = {0, 0, 0, 0};
        i32x8 z8 = {0, 0, 0, 0, 0, 0, 0, 0};
        __builtin_amdgcn_tensor_load_to_lds(g0, g1, z4, z4, z8, 0);
      }
#else
      const unsigned short* As = (const unsigned short*)Asrc;
      const uint4* p = (const uint4*)(As + (long)arow * lda + k0 + ac);
      *(uint4*)&alds[buf][ar * SA + ac]     = p[0];
      *(uint4*)&alds[buf][ar * SA + ac + 8] = p[1];
#endif
    } else {
      // per-lane async global->LDS copy, no conversion needed (bf16 source)
      const unsigned short* As = (const unsigned short*)Asrc;
      unsigned ldsa = (unsigned)(uintptr_t)&alds[buf][ar * SA + ac];
      unsigned long long gva = (unsigned long long)(uintptr_t)
                               (As + (long)arow * lda + k0 + ac);
      asm volatile("global_load_async_to_lds_b128 %0, %1, off\n\t"
                   "global_load_async_to_lds_b128 %0, %1, off offset:16"
                   :: "v"(ldsa), "v"(gva) : "memory");
    }
  };

  auto stageB = [&](int k0, int buf) {
    const float4* p = (const float4*)(Wg + (long)(k0 + bkr) * Ntotal + n0 + bnc);
    float4 f0 = p[0], f1 = p[1];
    float v[8] = { f0.x, f0.y, f0.z, f0.w, f1.x, f1.y, f1.z, f1.w };
    #pragma unroll
    for (int j = 0; j < 8; ++j) blds[buf][(bnc + j) * SB + bkr] = f2bf(v[j]);
  };

  v8f acc[2][2] = {};
  const int nk = KD / 32;

  stageA(0, 0);
  stageB(0, 0);

  for (int ki = 0; ki < nk; ++ki) {
    const int cur = ki & 1;
    if (ASTAGE == 1) {
#if __has_builtin(__builtin_amdgcn_tensor_load_to_lds)
      if (wave == 0) __builtin_amdgcn_s_wait_tensorcnt(0);
#endif
    }
    if (ASTAGE == 2) asm volatile("s_wait_asynccnt 0x0" ::: "memory");
    __syncthreads();                       // buffer `cur` staged & visible

    if (ki + 1 < nk) {                     // prefetch next tile into other buffer
      stageA((ki + 1) * 32, cur ^ 1);
      stageB((ki + 1) * 32, cur ^ 1);
    }

    Frag a0, a1, b0, b1;
    const unsigned short* ap0 = &alds[cur][(waveM +      lrow) * SA + akoff];
    const unsigned short* ap1 = &alds[cur][(waveM + 16 + lrow) * SA + akoff];
    a0.q[0] = *(const uint4*)ap0;  a0.q[1] = *(const uint4*)(ap0 + 16);
    a1.q[0] = *(const uint4*)ap1;  a1.q[1] = *(const uint4*)(ap1 + 16);
    const unsigned short* bp0 = &blds[cur][(waveN +      lrow) * SB + bkoff];
    const unsigned short* bp1 = &blds[cur][(waveN + 16 + lrow) * SB + bkoff];
    b0.q[0] = *(const uint4*)bp0;  b0.q[1] = *(const uint4*)(bp0 + 16);
    b1.q[0] = *(const uint4*)bp1;  b1.q[1] = *(const uint4*)(bp1 + 16);

    acc[0][0] = __builtin_amdgcn_wmma_f32_16x16x32_bf16(false, a0.v, false, b0.v, (short)0, acc[0][0], false, false);
    acc[0][1] = __builtin_amdgcn_wmma_f32_16x16x32_bf16(false, a0.v, false, b1.v, (short)0, acc[0][1], false, false);
    acc[1][0] = __builtin_amdgcn_wmma_f32_16x16x32_bf16(false, a1.v, false, b0.v, (short)0, acc[1][0], false, false);
    acc[1][1] = __builtin_amdgcn_wmma_f32_16x16x32_bf16(false, a1.v, false, b1.v, (short)0, acc[1][1], false, false);
  }

  // epilogue: bias + activation; C layout: lane<16 -> M=v, lane>=16 -> M=v+8, N=lane&15
  #pragma unroll
  for (int j = 0; j < 2; ++j) {
    const int col = n0 + waveN + j * 16 + lrow;
    const float bv = bg[col];
    #pragma unroll
    for (int i = 0; i < 2; ++i) {
      const int mbase = waveM + i * 16 + ((lane >> 4) << 3);
      #pragma unroll
      for (int v = 0; v < 8; ++v) {
        float x = acc[i][j][v] + bv;
        x = (ACT == 0) ? fmaxf(x, 0.f) : tanhf(x);
        const int oslot = slot0 + mbase + v;
        if (OUT_BF16) {
          ((unsigned short*)Out)[(long)oslot * ldo + col] = f2bf(x);
        } else {
          const int ob = scatterOut ? scatterOut[oslot] : oslot;
          if (ob >= 0) ((float*)Out)[(long)ob * ldo + col] = x;
        }
      }
    }
  }
}

static inline size_t alignup(size_t x) { return (x + 255) & ~(size_t)255; }

extern "C" void kernel_launch(void* const* d_in, const int* in_sizes, int n_in,
                              void* d_out, int out_size, void* d_ws, size_t ws_size,
                              hipStream_t stream) {
  (void)in_sizes; (void)n_in; (void)out_size; (void)ws_size;
  const float* state = (const float*)d_in[0];
  const int*   idx   = (const int*)  d_in[1];
  const float* W1    = (const float*)d_in[2];
  const float* b1    = (const float*)d_in[3];
  const float* W2    = (const float*)d_in[4];
  const float* b2    = (const float*)d_in[5];
  const float* W3    = (const float*)d_in[6];
  const float* b3    = (const float*)d_in[7];
  float* out = (float*)d_out;

  char* ws = (char*)d_ws;
  size_t off = 0;
  int* sched = (int*)(ws + off); off = alignup(off + NSLOT * sizeof(int));
  int* tgame = (int*)(ws + off); off = alignup(off + NT_TILES * sizeof(int));
  int* l3in  = (int*)(ws + off); off = alignup(off + NSLOT * sizeof(int));
  int* tpos  = (int*)(ws + off); off = alignup(off + BROWS * sizeof(int));
  unsigned short* h1 = (unsigned short*)(ws + off); off = alignup(off + (size_t)NSLOT * HID1 * 2);
  unsigned short* h2 = (unsigned short*)(ws + off); off = alignup(off + (size_t)NSLOT * HID2 * 2);

  build_schedule<<<1, 256, 0, stream>>>(idx, sched, tgame, l3in, tpos);

  // L1: routed [8192x512]x[512x1024], gather f32 state rows, relu, bf16 out
  gemm_wmma<DIN, 0, 0, true><<<dim3(NT_TILES, HID1 / BN), 256, 0, stream>>>(
      state, DIN, W1, (long)DIN * HID1, b1, HID1, h1, HID1, HID1, sched, tgame, nullptr);

  // L2: shared [9216x1024]x[1024x1024], TDM-staged bf16 A, relu, bf16 out
  gemm_wmma<HID1, 1, 0, true><<<dim3(NT_TILES, HID2 / BN), 256, 0, stream>>>(
      h1, HID1, W2, 0L, b2, 0, h2, HID2, HID2, nullptr, nullptr, nullptr);

  // L3: routed heads [9216x1024]x[1024x128], async-gathered bf16 A,
  //     bug-faithful gather/scatter, tanh, f32 out
  gemm_wmma<HID2, 2, 1, false><<<dim3(NT_TILES, AOUT / BN), 256, 0, stream>>>(
      h2, HID2, W3, (long)HID2 * AOUT, b3, AOUT, out, AOUT, AOUT, l3in, tgame, sched);
}
